// GEMBlock_62689342652488
// MI455X (gfx1250) — matrix operand order repaired and checked by reading, not compiled, not run on hardware
//
#include <hip/hip_runtime.h>
#include <hip/hip_bf16.h>

typedef __attribute__((ext_vector_type(16))) _Float16 v16h;
typedef __attribute__((ext_vector_type(8)))  float    v8f;

#define DIM 64
#define NV  50000
#define NE  800000
#define NB  5            // NUM_BASIS
#define KTOT 320         // NB * DIM
#define NTILES (NE / 16) // 50000 edge tiles
#define AP  328          // padded LDS row stride (halves), rows 16B-aligned
#define KP  72           // padded 64-wide LDS row stride (halves)

// Load a 16-element f16 fragment as two 16-byte LDS reads.
__device__ __forceinline__ v16h ld2q(const _Float16* p0, const _Float16* p1) {
    union { uint4 q[2]; v16h h; } u;
    u.q[0] = *(const uint4*)p0;
    u.q[1] = *(const uint4*)p1;
    return u.h;
}

// ---------------------------------------------------------------------------
// Kernel 1: accum[v][o] = sum_i x[v][i] * K_self[o][i]   (initializes accum)
// One block = 16 nodes (M), 4 waves = 4 N-tiles of 16, K = 64 (2 WMMA steps).
// ---------------------------------------------------------------------------
__global__ __launch_bounds__(128) void self_gemm_kernel(
        const float* __restrict__ x, const float* __restrict__ Ks,
        float* __restrict__ accum) {
    __shared__ alignas(16) _Float16 Kl[DIM][KP];   // Kl[o][i] = K_self[o][i] (B as [n][k])
    __shared__ alignas(16) _Float16 Xl[16][KP];    // A tile

    const int tid = threadIdx.x;
    for (int idx = tid; idx < DIM * DIM; idx += 128)
        Kl[idx >> 6][idx & 63] = (_Float16)Ks[idx];

    const int v0 = blockIdx.x * 16;
    {
        const int row = tid >> 3, c8 = tid & 7;
        const float4 a = *(const float4*)(x + (v0 + row) * DIM + c8 * 8);
        const float4 b = *(const float4*)(x + (v0 + row) * DIM + c8 * 8 + 4);
        _Float16* d = &Xl[row][c8 * 8];
        d[0] = (_Float16)a.x; d[1] = (_Float16)a.y; d[2] = (_Float16)a.z; d[3] = (_Float16)a.w;
        d[4] = (_Float16)b.x; d[5] = (_Float16)b.y; d[6] = (_Float16)b.z; d[7] = (_Float16)b.w;
    }
    __syncthreads();

    const int lane = tid & 31, wid = tid >> 5;
    const int n0 = wid * 16, nn = lane & 15, half = lane >> 4, mrow = lane & 15;

    v8f acc = {};
#pragma unroll
    for (int kk = 0; kk < 2; ++kk) {
        const int k0 = kk * 32;
        v16h af = ld2q(&Xl[mrow][k0 + 8 * half], &Xl[mrow][k0 + 16 + 8 * half]);
        v16h bf = ld2q(&Kl[n0 + nn][k0 + 8 * half], &Kl[n0 + nn][k0 + 16 + 8 * half]);
        acc = __builtin_amdgcn_wmma_f32_16x16x32_f16(false, af, false, bf,
                                                     (short)0, acc, false, false);
    }
#pragma unroll
    for (int r = 0; r < 8; ++r)
        accum[(v0 + r + 8 * half) * DIM + n0 + nn] = acc[r];
}

// ---------------------------------------------------------------------------
// Kernel 2: edge messages.  msg[e] = A[e,:] @ Wflat, scatter-add into accum.
//   A[e, b*64+i]    = basis[e,b] * transport(x[src[e]])[i]
//   Wflat[b*64+i,o] = W_neigh[b,o,i]
// Persistent blocks, 128 threads, 16 edges/tile, 4 waves x (K=320 WMMA chain).
// B fragments (loop-invariant) are hoisted into 10 v16h registers per lane;
// the W LDS region is then dead, so the A staging tile aliases the same LDS.
// ---------------------------------------------------------------------------
__global__ __launch_bounds__(128) void edge_msg_kernel(
        const float* __restrict__ x, const int* __restrict__ ei,
        const float* __restrict__ angles, const float* __restrict__ transp,
        const float* __restrict__ Wn, float* __restrict__ accum) {
    __shared__ alignas(16) _Float16 smem[DIM * AP];  // phase 1: Wl[o][k]; phase 2: Al[e][k]
    __shared__ int tgtL[16];

    const int tid = threadIdx.x;
    // Stage Wflat (transposed: row o, col k=b*64+i) into LDS as f16.
    for (int idx = tid; idx < DIM * KTOT; idx += 128) {
        const int o = idx / KTOT, c = idx % KTOT;
        const int b = c >> 6, i = c & 63;
        smem[o * AP + c] = (_Float16)Wn[b * (DIM * DIM) + o * DIM + i];
    }
    __syncthreads();

    const int lane = tid & 31, wid = tid >> 5;
    const int n0 = wid * 16, nn = lane & 15, half = lane >> 4, mrow = lane & 15;
    const int el = tid >> 3, c8 = tid & 7;   // staging role: edge-in-tile, 8-ch chunk

    // Hoist the loop-invariant B operand: 10 fragments (80 VGPRs/lane).
    v16h bf[10];
#pragma unroll
    for (int kk = 0; kk < 10; ++kk) {
        const _Float16* br = &smem[(n0 + nn) * AP + kk * 32 + 16 * half];
        bf[kk] = ld2q(br, br + 8);
    }

    for (int tile = blockIdx.x; tile < NTILES; tile += gridDim.x) {
        const int e0 = tile * 16;
        __syncthreads();   // all waves' LDS reads (bf or prev A-frags) complete
        {
            const int e = e0 + el;
            const int src = ei[e];
            if (c8 == 0) {
                tgtL[el] = ei[NE + e];
                // Prefetch next tile's metadata and gathered rows (global_prefetch_b8).
                const int tn = tile + gridDim.x;
                if (tn < NTILES) {
                    const int en = tn * 16 + el;
                    __builtin_prefetch(&ei[en], 0, 3);
                    __builtin_prefetch(&ei[NE + en], 0, 3);
                    __builtin_prefetch(&angles[en], 0, 3);
                    __builtin_prefetch(&transp[en], 0, 3);
                    __builtin_prefetch(x + (size_t)ei[en] * DIM, 0, 3);
                    __builtin_prefetch(x + (size_t)ei[en] * DIM + 32, 0, 3);
                }
            }
            const float ang = angles[e];
            float sa, ca; __sincosf(ang, &sa, &ca);
            const float basis[NB] = {1.f, ca, sa, ca * ca - sa * sa, 2.f * sa * ca};

            const float4 v0 = *(const float4*)(x + src * DIM + c8 * 8);
            const float4 v1 = *(const float4*)(x + src * DIM + c8 * 8 + 4);
            float vals[8] = {v0.x, v0.y, v0.z, v0.w, v1.x, v1.y, v1.z, v1.w};
            if (c8 >= 2) {                       // vector irreps: rotate pairs
                const float t = transp[e];
                const float phi = (c8 >= 6) ? 2.f * t : t;   // order 1 vs order 2
                float sp, cp; __sincosf(phi, &sp, &cp);
#pragma unroll
                for (int p = 0; p < 4; ++p) {
                    const float b0 = vals[2 * p], b1 = vals[2 * p + 1];
                    vals[2 * p]     = b0 * cp - b1 * sp;
                    vals[2 * p + 1] = b0 * sp + b1 * cp;
                }
            }
#pragma unroll
            for (int b = 0; b < NB; ++b) {
                _Float16* d = &smem[el * AP + b * DIM + c8 * 8];
#pragma unroll
                for (int q = 0; q < 8; ++q) d[q] = (_Float16)(vals[q] * basis[b]);
            }
        }
        __syncthreads();

        v8f acc = {};
#pragma unroll
        for (int kk = 0; kk < 10; ++kk) {
            const _Float16* ar = &smem[mrow * AP + kk * 32 + 16 * half];
            v16h af = ld2q(ar, ar + 8);
            acc = __builtin_amdgcn_wmma_f32_16x16x32_f16(false, af, false, bf[kk],
                                                         (short)0, acc, false, false);
        }
#pragma unroll
        for (int r = 0; r < 8; ++r) {
            const int m = r + 8 * half;
            atomicAdd(&accum[tgtL[m] * DIM + n0 + nn], acc[r]);
        }
    }
}

// ---------------------------------------------------------------------------
// Kernel 3: LayerNorm + equivariant nonlinearity + residual.
// One wave32 per node: lane holds channels {lane, lane+32}.
// ---------------------------------------------------------------------------
__global__ __launch_bounds__(256) void finalize_kernel(
        const float* __restrict__ accum, const float* __restrict__ x,
        const float* __restrict__ gamma, const float* __restrict__ beta,
        float* __restrict__ out) {
    const int lane = threadIdx.x & 31, wid = threadIdx.x >> 5;
    const int v = blockIdx.x * 8 + wid;
    if (v >= NV) return;

    float h0 = accum[v * DIM + lane];
    float h1 = accum[v * DIM + 32 + lane];
    float s = h0 + h1, s2 = h0 * h0 + h1 * h1;
#pragma unroll
    for (int off = 16; off >= 1; off >>= 1) {
        s  += __shfl_xor(s, off, 32);
        s2 += __shfl_xor(s2, off, 32);
    }
    const float mu  = s * (1.f / DIM);
    const float var = s2 * (1.f / DIM) - mu * mu;
    const float rs  = rsqrtf(var + 1e-5f);
    h0 = (h0 - mu) * rs * gamma[lane]      + beta[lane];
    h1 = (h1 - mu) * rs * gamma[lane + 32] + beta[lane + 32];

    // pair partner (adjacent channel) lives in lane^1 for both halves
    const float p0 = __shfl_xor(h0, 1, 32);
    const float p1 = __shfl_xor(h1, 1, 32);
    if (lane < 16) {
        h0 = fmaxf(h0, 0.f);                          // rho_0 scalars: ReLU
    } else {                                          // channels 16..31: order-1 pairs
        const float n  = fmaxf(sqrtf(h0 * h0 + p0 * p0), 1e-8f);
        const float sp = log1pf(expf(-fabsf(n))) + fmaxf(n, 0.f);   // softplus
        h0 *= sp / n;
    }
    {                                                 // channels 32..63: all vector pairs
        const float n  = fmaxf(sqrtf(h1 * h1 + p1 * p1), 1e-8f);
        const float sp = log1pf(expf(-fabsf(n))) + fmaxf(n, 0.f);
        h1 *= sp / n;
    }
    out[v * DIM + lane]      = h0 + x[v * DIM + lane];
    out[v * DIM + 32 + lane] = h1 + x[v * DIM + 32 + lane];
}

extern "C" void kernel_launch(void* const* d_in, const int* in_sizes, int n_in,
                              void* d_out, int out_size, void* d_ws, size_t ws_size,
                              hipStream_t stream) {
    const float* x      = (const float*)d_in[0];
    const int*   ei     = (const int*)  d_in[1];   // [0:E]=src, [E:2E]=tgt
    const float* ang    = (const float*)d_in[2];
    const float* trn    = (const float*)d_in[3];
    const float* Ks     = (const float*)d_in[4];
    const float* Wn     = (const float*)d_in[5];
    const float* gamma  = (const float*)d_in[6];
    const float* beta   = (const float*)d_in[7];
    float*       out    = (float*)d_out;
    float*       accum  = (float*)d_ws;            // V*DIM floats of scratch

    self_gemm_kernel<<<NV / 16, 128, 0, stream>>>(x, Ks, accum);
    edge_msg_kernel<<<1024, 128, 0, stream>>>(x, ei, ang, trn, Wn, accum);
    finalize_kernel<<<NV / 8, 256, 0, stream>>>(accum, x, gamma, beta, out);
}
